// KerasModelGRU4REC_71906342469708
// MI455X (gfx1250) — matrix operand order repaired and checked by reading, not compiled, not run on hardware
//
#include <hip/hip_runtime.h>
#include <hip/hip_bf16.h>

// Problem constants (from reference)
#define BATCH   512
#define SEQLEN  256
#define EMB     128
#define HID     128
#define N3      384      // 3*HID
#define TROWS   16       // batch rows per block (one WMMA M-tile)
#define BLOCK   384      // 12 waves of 32
#define TPW     2        // N-tiles per wave: 24 tiles / 12 waves

typedef __attribute__((ext_vector_type(16))) __bf16 v16bf;
typedef __attribute__((ext_vector_type(8)))  float  v8f;

union BfFrag { v16bf v; unsigned short s[16]; };

__device__ __forceinline__ unsigned short f2bf(float f) {
    // native f32->bf16 (RNE); lowers to v_cvt_*bf16* when the target has it,
    // otherwise clang expands the equivalent RNE integer sequence.
    union { __bf16 h; unsigned short u; } cv;
    cv.h = (__bf16)f;
    return cv.u;
}

__device__ __forceinline__ float sigmoidf_(float x) {
    return 1.0f / (1.0f + __expf(-x));
}

__device__ __forceinline__ float tanhf_(float x) {
    const float e = __expf(-2.0f * x);
    return (1.0f - e) / (1.0f + e);
}

// ---------------------------------------------------------------------------
// Persistent GRU kernel: one block owns 16 batch rows for all 256 time steps.
//  * W/U: bf16 register fragments (128 VGPRs/wave), resident for the kernel.
//  * h:   f32 master + bf16 mirror in LDS (mirror written once per element
//         in the gate phase -> GEMM A-path is pure ds_load_b128).
//  * Embedding gather is software-pipelined one full step ahead so the HBM
//    latency hides behind the 16 WMMAs + gate math of the current step.
// WMMA: v_wmma_f32_16x16x32_bf16, f32 accumulate.
// ---------------------------------------------------------------------------
__global__ __launch_bounds__(BLOCK) void gru_persistent_kernel(
    const int*   __restrict__ mid_hist,   // [BATCH, SEQLEN]
    const int*   __restrict__ mask,       // [BATCH, SEQLEN]
    const float* __restrict__ emb_table,  // [N_MID, EMB]
    const float* __restrict__ W,          // [EMB, N3] row-major
    const float* __restrict__ U,          // [HID, N3] row-major
    const float* __restrict__ bias,       // [2, N3]
    float*       __restrict__ user_vec)   // [BATCH, HID]
{
    __shared__ int            s_idx[2][TROWS];    // double-buffered indices
    __shared__ int            s_msk[2][TROWS];
    __shared__ unsigned short s_x [TROWS * EMB];  // gathered embeddings, bf16
    __shared__ unsigned short s_hb[TROWS * HID];  // hidden state, bf16 mirror
    __shared__ float          s_h [TROWS * HID];  // hidden state, f32 master
    __shared__ float          s_zr[TROWS * 256];  // (x@W + h@U) cols 0..255
    __shared__ float          s_xh[TROWS * HID];  // (x@W) cols 256..383
    __shared__ float          s_rh[TROWS * HID];  // (h@U) cols 256..383
    __shared__ float          s_bz [HID];         // b_in[z]+b_rec[z]
    __shared__ float          s_br [HID];         // b_in[r]+b_rec[r]
    __shared__ float          s_bxh[HID];         // b_in[h]
    __shared__ float          s_brh[HID];         // b_rec[h]

    const int tid  = threadIdx.x;
    const int lane = tid & 31;
    // wave id as an SGPR: scalar branches + scalar weight addressing
    const int wv   = __builtin_amdgcn_readfirstlane(tid >> 5);
    const int b0   = blockIdx.x * TROWS;
    const int n    = lane & 15;   // column within 16-wide N-tile (B/C/D layout)
    const int g    = lane >> 4;   // lane-group selects K/M halves
    const int m    = lane & 15;   // A-matrix row

    // ---- Preload weight fragments (bf16 B-operand layout) ------------------
    // B (KxN) striping: VGPR v holds K = 32c + 16g + 2v, 2v+1 for column n.
    BfFrag wf[TPW][4];
    BfFrag uf[TPW][4];
    for (int jt = 0; jt < TPW; ++jt) {
        const int cb = 16 * (wv * TPW + jt);
        for (int c = 0; c < 4; ++c) {
            #pragma unroll
            for (int v = 0; v < 8; ++v) {
                const int k = 32 * c + 16 * g + 2 * v;
                wf[jt][c].s[2*v]   = f2bf(W[(size_t)k       * N3 + cb + n]);
                wf[jt][c].s[2*v+1] = f2bf(W[(size_t)(k + 1) * N3 + cb + n]);
                uf[jt][c].s[2*v]   = f2bf(U[(size_t)k       * N3 + cb + n]);
                uf[jt][c].s[2*v+1] = f2bf(U[(size_t)(k + 1) * N3 + cb + n]);
            }
        }
    }

    // ---- h0 = 0 (both copies), folded biases, t=0 indices ------------------
    for (int i = tid; i < TROWS * HID; i += BLOCK) { s_h[i] = 0.0f; s_hb[i] = 0; }
    if (tid < HID) {
        s_bz [tid] = bias[tid]       + bias[N3 + tid];
        s_br [tid] = bias[128 + tid] + bias[N3 + 128 + tid];
        s_bxh[tid] = bias[256 + tid];
        s_brh[tid] = bias[N3 + 256 + tid];
    }
    if (tid < TROWS) {
        s_idx[0][tid] = mid_hist[(size_t)(b0 + tid) * SEQLEN];
        s_msk[0][tid] = mask    [(size_t)(b0 + tid) * SEQLEN];
    }
    __syncthreads();

    // ---- gather prefetch lanes: 512 float4 chunks over 384 threads ---------
    const int q0 = tid,         row0 = q0 >> 5, c40 = q0 & 31;
    const int q1 = tid + BLOCK, row1 = q1 >> 5, c41 = q1 & 31;  // tid<128 only
    float4 pf0, pf1 = make_float4(0.f, 0.f, 0.f, 0.f);
    pf0 = ((const float4*)(emb_table + (size_t)s_idx[0][row0] * EMB))[c40];
    if (tid < 2 * TROWS * 32 - BLOCK)
        pf1 = ((const float4*)(emb_table + (size_t)s_idx[0][row1] * EMB))[c41];

    for (int t = 0; t < SEQLEN; ++t) {
        const int par = t & 1;

        // -- phase A: commit prefetched embeddings (f32->bf16) to LDS,
        //    and fetch indices/mask for step t+1 --
        {
            int o = row0 * EMB + c40 * 4;
            s_x[o + 0] = f2bf(pf0.x); s_x[o + 1] = f2bf(pf0.y);
            s_x[o + 2] = f2bf(pf0.z); s_x[o + 3] = f2bf(pf0.w);
            if (tid < 2 * TROWS * 32 - BLOCK) {
                o = row1 * EMB + c41 * 4;
                s_x[o + 0] = f2bf(pf1.x); s_x[o + 1] = f2bf(pf1.y);
                s_x[o + 2] = f2bf(pf1.z); s_x[o + 3] = f2bf(pf1.w);
            }
        }
        const int tn = (t + 1 < SEQLEN) ? (t + 1) : t;
        if (tid < TROWS) {
            s_idx[par ^ 1][tid] = mid_hist[(size_t)(b0 + tid) * SEQLEN + tn];
            s_msk[par ^ 1][tid] = mask    [(size_t)(b0 + tid) * SEQLEN + tn];
        }
        __syncthreads();   // s_x ready for step t; s_idx[t+1] visible

        // -- issue t+1 embedding gather now; latency hides behind this step --
        pf0 = ((const float4*)(emb_table
                               + (size_t)s_idx[par ^ 1][row0] * EMB))[c40];
        if (tid < 2 * TROWS * 32 - BLOCK)
            pf1 = ((const float4*)(emb_table
                                   + (size_t)s_idx[par ^ 1][row1] * EMB))[c41];

        // -- dual GEMM: x-path (s_x @ W-tiles) + rec-path (s_hb @ U-tiles) --
        v8f accx[TPW] = {};
        v8f acch[TPW] = {};
        #pragma unroll
        for (int c = 0; c < 4; ++c) {
            BfFrag ax, ah;  // A k order used here (same as B): g=0 -> K {0..15}, g=1 -> {16..31}
            #pragma unroll
            for (int v = 0; v < 8; ++v) {
                const int k0 = 16 * g + 2 * v;
                const int kk = 32 * c + k0;
                ax.s[2*v]   = s_x [m * EMB + kk];
                ax.s[2*v+1] = s_x [m * EMB + kk + 1];
                ah.s[2*v]   = s_hb[m * HID + kk];
                ah.s[2*v+1] = s_hb[m * HID + kk + 1];
            }
            #pragma unroll
            for (int jt = 0; jt < TPW; ++jt) {
                accx[jt] = __builtin_amdgcn_wmma_f32_16x16x32_bf16(
                    false, ax.v, false, wf[jt][c].v, (short)0, accx[jt],
                    false, false);
                acch[jt] = __builtin_amdgcn_wmma_f32_16x16x32_bf16(
                    false, ah.v, false, uf[jt][c].v, (short)0, acch[jt],
                    false, false);
            }
        }

        // -- spill pre-activations (C/D layout: VGPR v -> row v + 8g) --
        // waves 0..7 own tiles 0..15 (z,r): only the x+rec SUM is needed
        // waves 8..11 own tiles 16..23 (candidate): xh and rh kept separate
        if (wv < 8) {           // scalar branch (wv is SGPR)
            #pragma unroll
            for (int jt = 0; jt < TPW; ++jt) {
                const int cb = 16 * (wv * TPW + jt);
                #pragma unroll
                for (int v = 0; v < 8; ++v) {
                    const int row = v + 8 * g;
                    s_zr[row * 256 + cb + n] = accx[jt][v] + acch[jt][v];
                }
            }
        } else {
            #pragma unroll
            for (int jt = 0; jt < TPW; ++jt) {
                const int cb2 = 16 * (wv * TPW + jt) - 256;
                #pragma unroll
                for (int v = 0; v < 8; ++v) {
                    const int row = v + 8 * g;
                    s_xh[row * HID + cb2 + n] = accx[jt][v];
                    s_rh[row * HID + cb2 + n] = acch[jt][v];
                }
            }
        }
        __syncthreads();

        // -- elementwise masked GRU gate update (f32 master, bf16 mirror) --
        for (int e = tid; e < TROWS * HID; e += BLOCK) {
            const int mr = e >> 7, j = e & 127;
            const float z  = sigmoidf_(s_zr[mr * 256 + j]       + s_bz[j]);
            const float r  = sigmoidf_(s_zr[mr * 256 + 128 + j] + s_br[j]);
            const float y  = s_xh[mr * HID + j] + s_bxh[j]
                             + r * (s_rh[mr * HID + j] + s_brh[j]);
            const float hh = tanhf_(y);
            const float ho = s_h[e];
            const float hn = z * ho + (1.0f - z) * hh;
            const float hw = s_msk[par][mr] ? hn : ho;
            s_h[e]  = hw;
            s_hb[e] = f2bf(hw);
        }
        __syncthreads();
    }

    // -- write user_vec --
    for (int e = tid; e < TROWS * HID; e += BLOCK)
        user_vec[(size_t)(b0 + (e >> 7)) * HID + (e & 127)] = s_h[e];
}

// ---------------------------------------------------------------------------
// item_emb = emb_table[mid]  (trivial gather)
// ---------------------------------------------------------------------------
__global__ void item_gather_kernel(const int* __restrict__ mid,
                                   const float* __restrict__ emb_table,
                                   float* __restrict__ out_item)
{
    const int i = blockIdx.x * blockDim.x + threadIdx.x;
    if (i < BATCH * EMB) {
        const int bb = i >> 7;
        out_item[i] = emb_table[(size_t)mid[bb] * EMB + (i & 127)];
    }
}

extern "C" void kernel_launch(void* const* d_in, const int* in_sizes, int n_in,
                              void* d_out, int out_size, void* d_ws, size_t ws_size,
                              hipStream_t stream)
{
    (void)in_sizes; (void)n_in; (void)out_size; (void)d_ws; (void)ws_size;
    const int*   mid      = (const int*)  d_in[0];
    const int*   mid_hist = (const int*)  d_in[1];
    const int*   mask     = (const int*)  d_in[2];
    const float* emb      = (const float*)d_in[3];
    const float* W        = (const float*)d_in[4];
    const float* U        = (const float*)d_in[5];
    const float* b        = (const float*)d_in[6];
    float* out = (float*)d_out;                 // [user_vec | item_emb]

    item_gather_kernel<<<(BATCH * EMB + 255) / 256, 256, 0, stream>>>(
        mid, emb, out + (size_t)BATCH * HID);

    gru_persistent_kernel<<<BATCH / TROWS, BLOCK, 0, stream>>>(
        mid_hist, mask, emb, W, U, b, out);
}
